// EncoderLayer_25211458027663
// MI455X (gfx1250) — matrix operand order repaired and checked by reading, not compiled, not run on hardware
//
#include <hip/hip_runtime.h>
#include <hip/hip_bf16.h>
#include <math.h>

// ---------------------------------------------------------------------------
// Problem constants (from reference)
// ---------------------------------------------------------------------------
#define NB   2
#define NN   2048
#define NK   36
#define HD   128
#define NIN  256
#define DIN  384            // H + NIN
#define NH4  4
#define CH16 16
#define PQ4  4
#define PV8  8

static constexpr int    BN   = NB * NN;        // 4096 nodes
static constexpr int    BNK  = BN * NK;        // 147456 edges

// ---------------------------------------------------------------------------
// WMMA types (CDNA5 / gfx1250, wave32)
// ---------------------------------------------------------------------------
typedef __attribute__((ext_vector_type(16))) __bf16 v16bf;
typedef __attribute__((ext_vector_type(8)))  float  v8f;

union Frag16 { v16bf v; uint4 q[2]; };         // 16 bf16 = two 16B chunks
union Pack2  { unsigned u; __bf16 h[2]; };
union Pack4  { uint2 u2;  __bf16 h[4]; };

__device__ __forceinline__ float softplusf(float x) {
    return (x > 0.0f) ? (x + log1pf(expf(-x))) : log1pf(expf(x));
}

// ---------------------------------------------------------------------------
// Tiled GEMM:  C[M,N] = act( A[M,K] @ W[K,N] + bias[N] )
//
// Block: 256 threads (8 waves, wave32). Block tile: 128 x (NT*16).
// K-panel depth 64 staged through LDS *as bf16* (A: 128x64 = 16KB, B:
// 64xNTC <= 16KB).  fp32->bf16 conversion happens once at staging; the inner
// loop is pure ds_load_b128 + v_wmma_f32_16x16x32_bf16:
//   - A panel row-major: each lane's fragment = two contiguous 16B runs
//     (K0..7 / K16..23 with half-wave offset +8)  -> 2x ds_load_b128.
//   - B panel stored PRE-SWIZZLED into fragment order: for each
//     (k-half, col-tile, lane) the 8 K-pair words are contiguous (32B)
//     -> 2x ds_load_b128, zero register shuffles.
// Each wave reuses its A fragment across NT column tiles.
//
// Requirements (all call sites satisfy): M%128==0, K%64==0, N%(NT*16)==0.
// act: 0 = none, 1 = exact GELU, 2 = ReLU
// ---------------------------------------------------------------------------
template <int NT>
__global__ void __launch_bounds__(256)
gemm_wmma_tile_kernel(const float* __restrict__ A, const float* __restrict__ W,
                      const float* __restrict__ bias, float* __restrict__ C,
                      int M, int Kdim, int Ndim, int act)
{
    constexpr int NTC = NT * 16;
    __shared__ __bf16   ldsA[128 * 64];        // [row][k]
    __shared__ unsigned ldsB[32 * NTC];        // fragment-order K-pair words

    const int t    = threadIdx.x;
    const int lane = t & 31;
    const int wave = t >> 5;
    const int row0 = blockIdx.x * 128;
    const int colb = blockIdx.y * NTC;
    const int r    = lane & 15;          // row-in-tile for A, col-in-tile for B
    const int hi   = lane >> 4;          // half-wave selector

    v8f acc[NT];
#pragma unroll
    for (int i = 0; i < NT; ++i) acc[i] = (v8f){};

    for (int kk = 0; kk < Kdim; kk += 64) {
        __syncthreads();
        // ---- stage A panel: 128 rows x 64 k, float4 loads, bf16 packed ----
#pragma unroll
        for (int i = 0; i < 8; ++i) {
            int idx = t * 8 + i;                 // float4 index 0..2047
            int row = idx >> 4;                  // 16 float4 per row
            int qd  = idx & 15;
            float4 av = *(const float4*)(A + (size_t)(row0 + row) * Kdim + kk + qd * 4);
            Pack4 pk;
            pk.h[0] = (__bf16)av.x; pk.h[1] = (__bf16)av.y;
            pk.h[2] = (__bf16)av.z; pk.h[3] = (__bf16)av.w;
            *(uint2*)&ldsA[row * 64 + qd * 4] = pk.u2;
        }
        // ---- stage B panel: coalesced global reads, swizzled LDS stores ----
        // word (k2, cc) = bf16 pair (W[kk+2k2][colb+cc], W[kk+2k2+1][colb+cc])
        // placed so that each (k-half, col-tile, lane) owns 8 contiguous words.
        for (int i = t; i < 32 * NTC; i += 256) {
            int k2 = i / NTC, cc = i % NTC;
            int k  = k2 * 2;                     // 0..62, even
            Pack2 pk;
            pk.h[0] = (__bf16)W[(size_t)(kk + k)     * Ndim + colb + cc];
            pk.h[1] = (__bf16)W[(size_t)(kk + k + 1) * Ndim + colb + cc];
            int ksh  = k >> 5;                   // which 32-deep half
            int hi2  = (k >> 3) & 1;             // half-wave of this k
            int j    = ((k & 7) >> 1) + ((k >> 4) & 1) * 4;   // word index within fragment
            int dst  = (((ksh * NT) + (cc >> 4)) * 32 + hi2 * 16 + (cc & 15)) * 8 + j;
            ldsB[dst] = pk.u;
        }
        // speculative prefetch of next A K-panel (global_prefetch_b8)
        if (kk + 64 < Kdim)
            __builtin_prefetch(A + (size_t)(row0 + (t >> 1)) * Kdim + kk + 64 + (t & 1) * 32, 0, 1);
        __syncthreads();

#pragma unroll
        for (int ks = 0; ks < 64; ks += 32) {
            // A 16x32 bf16 fragment: two contiguous 16B runs per lane
            Frag16 af;
            const __bf16* abase = &ldsA[(wave * 16 + r) * 64 + ks + hi * 8];
            af.q[0] = *(const uint4*)(abase);
            af.q[1] = *(const uint4*)(abase + 16);
#pragma unroll
            for (int ct = 0; ct < NT; ++ct) {
                Frag16 bfv;
                const unsigned* bbase =
                    &ldsB[((((ks >> 5) * NT) + ct) * 32 + lane) * 8];
                bfv.q[0] = *(const uint4*)(bbase);
                bfv.q[1] = *(const uint4*)(bbase + 4);
                acc[ct] = __builtin_amdgcn_wmma_f32_16x16x32_bf16(
                              false, af.v, false, bfv.v, (short)0, acc[ct], false, false);
            }
        }
    }

    // ---- epilogue: bias + activation, fp32 store ----
#pragma unroll
    for (int ct = 0; ct < NT; ++ct) {
        int col = colb + ct * 16 + r;
        float bv = bias[col];
#pragma unroll
        for (int rr = 0; rr < 8; ++rr) {
            int row = row0 + wave * 16 + rr + 8 * hi;   // C frag: VGPR rr -> M
            float v = acc[ct][rr] + bv;
            if (act == 1)      v = 0.5f * v * (1.0f + erff(v * 0.70710678118654752f));
            else if (act == 2) v = fmaxf(v, 0.0f);
            C[(size_t)row * Ndim + col] = v;
        }
    }
}

// ---------------------------------------------------------------------------
// h_EV[m, 0:384] = [ h_V(center) | h_E(m) | h_V(neighbor) ]
// ---------------------------------------------------------------------------
__global__ void build_hEV_kernel(const float* __restrict__ hV,
                                 const float* __restrict__ hE,
                                 const int*   __restrict__ E_idx,
                                 float* __restrict__ out)
{
    int m = blockIdx.x;                 // 0..BNK-1
    int t = threadIdx.x;                // 0..383
    int b = m / (NN * NK);
    int n = (m / NK) % NN;
    float v;
    if (t < 128) {
        v = hV[(size_t)(b * NN + n) * HD + t];
    } else if (t < 256) {
        v = hE[(size_t)m * HD + (t - 128)];
    } else {
        int j = E_idx[m];
        v = hV[(size_t)(b * NN + j) * HD + (t - 256)];
    }
    out[(size_t)m * DIN + t] = v;
}

// ---------------------------------------------------------------------------
// h_V' = LN( h_V + (1/36) * sum_k mask_att * msg , g, b ) ; block per node
// ---------------------------------------------------------------------------
__global__ void node_update_kernel(const float* __restrict__ hV,
                                   const float* __restrict__ msg,
                                   const float* __restrict__ mask_att,
                                   const float* __restrict__ g,
                                   const float* __restrict__ bvec,
                                   float* __restrict__ out)
{
    __shared__ float red[HD];
    int node = blockIdx.x;
    int c    = threadIdx.x;
    float s = 0.f;
    for (int k = 0; k < NK; ++k)
        s += mask_att[node * NK + k] * msg[(size_t)(node * NK + k) * HD + c];
    float x = hV[(size_t)node * HD + c] + s * (1.0f / 36.0f);

    red[c] = x; __syncthreads();
    for (int off = 64; off > 0; off >>= 1) { if (c < off) red[c] += red[c + off]; __syncthreads(); }
    float mean = red[0] * (1.0f / HD); __syncthreads();
    float d = x - mean;
    red[c] = d * d; __syncthreads();
    for (int off = 64; off > 0; off >>= 1) { if (c < off) red[c] += red[c + off]; __syncthreads(); }
    float var = red[0] * (1.0f / HD);
    out[(size_t)node * HD + c] = d * rsqrtf(var + 1e-5f) * g[c] + bvec[c];
}

// ---------------------------------------------------------------------------
// out = (rowmask ? rowmask[row] : 1) * LN(x + y) ; block per row (HD channels)
// ---------------------------------------------------------------------------
__global__ void add_ln_kernel(const float* __restrict__ x,
                              const float* __restrict__ y,
                              const float* __restrict__ g,
                              const float* __restrict__ bvec,
                              const float* __restrict__ rowmask,
                              float* __restrict__ out)
{
    __shared__ float red[HD];
    long row = blockIdx.x;
    int  c   = threadIdx.x;
    float v = x[row * HD + c] + y[row * HD + c];
    red[c] = v; __syncthreads();
    for (int off = 64; off > 0; off >>= 1) { if (c < off) red[c] += red[c + off]; __syncthreads(); }
    float mean = red[0] * (1.0f / HD); __syncthreads();
    float d = v - mean;
    red[c] = d * d; __syncthreads();
    for (int off = 64; off > 0; off >>= 1) { if (c < off) red[c] += red[c + off]; __syncthreads(); }
    float var = red[0] * (1.0f / HD);
    float o = d * rsqrtf(var + 1e-5f) * g[c] + bvec[c];
    if (rowmask) o *= rowmask[row];
    out[row * HD + c] = o;
}

// ---------------------------------------------------------------------------
// Gather per-edge frames R_j (3x3) and t_j (3)
// ---------------------------------------------------------------------------
__global__ void gather_frames_kernel(const int* __restrict__ E_idx,
                                     const float* __restrict__ Rrot,
                                     const float* __restrict__ Rtrans,
                                     float* __restrict__ Rj, float* __restrict__ tj)
{
    int e = blockIdx.x * blockDim.x + threadIdx.x;
    if (e >= BNK) return;
    int b   = e / (NN * NK);
    int src = b * NN + E_idx[e];
    for (int i = 0; i < 9; ++i) Rj[(size_t)e * 9 + i] = Rrot[(size_t)src * 9 + i];
    for (int i = 0; i < 3; ++i) tj[(size_t)e * 3 + i] = Rtrans[(size_t)src * 3 + i];
}

// ---------------------------------------------------------------------------
// In-place: pt <- R_j @ pt + t_j   (npts 3-vectors per edge)
// ---------------------------------------------------------------------------
__global__ void apply_rot_kernel(float* __restrict__ pts,
                                 const float* __restrict__ Rj,
                                 const float* __restrict__ tj, int npts)
{
    long id  = blockIdx.x * (long)blockDim.x + threadIdx.x;
    long tot = (long)BNK * npts;
    if (id >= tot) return;
    long e = id / npts;
    float* p = pts + id * 3;
    const float* R = Rj + e * 9;
    const float* t = tj + e * 3;
    float x = p[0], y = p[1], z = p[2];
    p[0] = R[0] * x + R[1] * y + R[2] * z + t[0];
    p[1] = R[3] * x + R[4] * y + R[5] * z + t[1];
    p[2] = R[6] * x + R[7] * y + R[8] * z + t[2];
}

// ---------------------------------------------------------------------------
// IPA attention core. Thread per (m, h, q). Writes feats[BNK,192]:
//   [ o (64) | local (96) | |local| (32) ]
// ---------------------------------------------------------------------------
__global__ void ipa_attn_kernel(const float* __restrict__ q,
                                const float* __restrict__ kv,
                                const float* __restrict__ qp,
                                const float* __restrict__ kvp,
                                const float* __restrict__ Rj,
                                const float* __restrict__ tj,
                                const float* __restrict__ head_w,
                                const float* __restrict__ mask_j,
                                float* __restrict__ feats)
{
    long id  = blockIdx.x * (long)blockDim.x + threadIdx.x;
    long tot = (long)BN * NH4 * NK;
    if (id >= tot) return;
    int qi = (int)(id % NK);
    int h  = (int)((id / NK) % NH4);
    long m = id / ((long)NK * NH4);
    long eq = m * NK + qi;

    float qv[CH16];
    for (int c = 0; c < CH16; ++c) qv[c] = q[eq * 64 + h * CH16 + c];
    float qpts[PQ4][3]; float qn = 0.f;
    for (int p = 0; p < PQ4; ++p)
        for (int a = 0; a < 3; ++a) {
            float v = qp[eq * 48 + (h * PQ4 + p) * 3 + a];
            qpts[p][a] = v; qn += v * v;
        }

    const float hw = softplusf(head_w[h]);
    const float sc = 0.14433756729740644f;   // sqrt(1/(3*16))
    const float wc = 0.13608276348795434f;   // sqrt(1/(3*4*4.5))

    float att[NK];
    float mx = -1e30f;
    for (int k = 0; k < NK; ++k) {
        long ek = m * NK + k;
        float dot = 0.f;
        for (int c = 0; c < CH16; ++c) dot += qv[c] * kv[ek * 128 + h * 32 + c];
        float kn = 0.f, cross = 0.f;
        for (int p = 0; p < PQ4; ++p)
            for (int a = 0; a < 3; ++a) {
                float kp = kvp[ek * 144 + h * 36 + p * 3 + a];
                kn += kp * kp; cross += qpts[p][a] * kp;
            }
        float d2 = qn + kn - 2.f * cross;
        float l  = dot * sc - 0.5f * wc * hw * d2 + (mask_j[ek] - 1.0f) * 1e9f;
        att[k] = l; mx = fmaxf(mx, l);
    }
    float denom = 0.f;
    for (int k = 0; k < NK; ++k) { float e = expf(att[k] - mx); att[k] = e; denom += e; }
    float inv = 1.0f / denom;

    float* frow = feats + eq * 192;
    for (int c = 0; c < CH16; ++c) {
        float s = 0.f;
        for (int k = 0; k < NK; ++k) s += att[k] * kv[(m * NK + k) * 128 + h * 32 + 16 + c];
        frow[h * CH16 + c] = s * inv;
    }
    const float* Rq = Rj + eq * 9;
    const float* tq = tj + eq * 3;
    for (int p = 0; p < PV8; ++p) {
        float o0 = 0.f, o1 = 0.f, o2 = 0.f;
        for (int k = 0; k < NK; ++k) {
            const float* vp = kvp + (m * NK + k) * 144 + h * 36 + 12 + p * 3;
            float w = att[k];
            o0 += w * vp[0]; o1 += w * vp[1]; o2 += w * vp[2];
        }
        o0 = o0 * inv - tq[0]; o1 = o1 * inv - tq[1]; o2 = o2 * inv - tq[2];
        float la[3];                        // R^T @ o   ('mqba' einsum)
        for (int a = 0; a < 3; ++a)
            la[a] = Rq[0 * 3 + a] * o0 + Rq[1 * 3 + a] * o1 + Rq[2 * 3 + a] * o2;
        frow[64 + (h * PV8 + p) * 3 + 0] = la[0];
        frow[64 + (h * PV8 + p) * 3 + 1] = la[1];
        frow[64 + (h * PV8 + p) * 3 + 2] = la[2];
        frow[160 + h * PV8 + p] =
            sqrtf(la[0] * la[0] + la[1] * la[1] + la[2] * la[2] + 1e-8f);
    }
}

// ---------------------------------------------------------------------------
// Small utility kernels
// ---------------------------------------------------------------------------
__global__ void zero_kernel(float* __restrict__ p, long n)
{
    long i = blockIdx.x * (long)blockDim.x + threadIdx.x;
    if (i < n) p[i] = 0.f;
}

__global__ void add_elem_kernel(const float* __restrict__ a,
                                const float* __restrict__ b,
                                float* __restrict__ o, long n)
{
    long i = blockIdx.x * (long)blockDim.x + threadIdx.x;
    if (i < n) o[i] = a[i] + b[i];
}

__global__ void seg_accum_kernel(const int* __restrict__ E_idx,
                                 const float* __restrict__ mask_j,
                                 const float* __restrict__ ipa_out,
                                 float* __restrict__ sums,
                                 float* __restrict__ counts)
{
    long id = blockIdx.x * (long)blockDim.x + threadIdx.x;
    if (id >= (long)BNK * HD) return;
    long e = id >> 7;
    int  c = (int)(id & 127);
    int  b = (int)(e / (NN * NK));
    int  tgt = b * NN + E_idx[e];
    atomicAdd(&sums[(size_t)tgt * HD + c], mask_j[e] * ipa_out[e * HD + c]);
    if (c == 0) atomicAdd(&counts[tgt], 1.0f);
}

__global__ void seg_ln_kernel(const float* __restrict__ sums,
                              const float* __restrict__ counts,
                              const float* __restrict__ g,
                              const float* __restrict__ bvec,
                              float* __restrict__ ne)
{
    __shared__ float red[HD];
    int node = blockIdx.x;
    int c    = threadIdx.x;
    float cnt = counts[node];
    if (cnt == 0.f) cnt = 1.0f;
    float x = sums[(size_t)node * HD + c] / cnt;
    red[c] = x; __syncthreads();
    for (int off = 64; off > 0; off >>= 1) { if (c < off) red[c] += red[c + off]; __syncthreads(); }
    float mean = red[0] * (1.0f / HD); __syncthreads();
    float d = x - mean;
    red[c] = d * d; __syncthreads();
    for (int off = 64; off > 0; off >>= 1) { if (c < off) red[c] += red[c + off]; __syncthreads(); }
    float var = red[0] * (1.0f / HD);
    ne[(size_t)node * HD + c] = d * rsqrtf(var + 1e-5f) * g[c] + bvec[c];
}

// uv = ((ne2 * um) @ bb_w + bb_b) * um ;  um = mask_V*(1-fixed); N=6 -> scalar
__global__ void bb_kernel(const float* __restrict__ ne2,
                          const float* __restrict__ maskV,
                          const float* __restrict__ fixedm,
                          const float* __restrict__ w,
                          const float* __restrict__ bvec,
                          float* __restrict__ uv)
{
    int id = blockIdx.x * blockDim.x + threadIdx.x;
    if (id >= BN * 6) return;
    int node = id / 6, j = id % 6;
    float um = maskV[node] * (1.0f - fixedm[node]);
    float s = bvec[j];
    for (int c = 0; c < HD; ++c)
        s += ne2[(size_t)node * HD + c] * um * w[c * 6 + j];
    uv[node * 6 + j] = s * um;
}

// quaternion -> rotation update; new frames written to output buffers
__global__ void finalize_kernel(const float* __restrict__ uv,
                                const float* __restrict__ Rrot,
                                const float* __restrict__ Rtrans,
                                float* __restrict__ out_rot,
                                float* __restrict__ out_trans)
{
    int node = blockIdx.x * blockDim.x + threadIdx.x;
    if (node >= BN) return;
    float w = 1.0f, x = uv[node * 6 + 0], y = uv[node * 6 + 1], z = uv[node * 6 + 2];
    float inv = rsqrtf(w * w + x * x + y * y + z * z);
    w *= inv; x *= inv; y *= inv; z *= inv;
    float ru[9] = { 1 - 2 * (y * y + z * z), 2 * (x * y - w * z), 2 * (x * z + w * y),
                    2 * (x * y + w * z), 1 - 2 * (x * x + z * z), 2 * (y * z - w * x),
                    2 * (x * z - w * y), 2 * (y * z + w * x), 1 - 2 * (x * x + y * y) };
    const float* R = Rrot + (size_t)node * 9;
    for (int i = 0; i < 3; ++i)
        for (int j = 0; j < 3; ++j) {
            float s = 0.f;
            for (int k = 0; k < 3; ++k) s += R[i * 3 + k] * ru[k * 3 + j];
            out_rot[(size_t)node * 9 + i * 3 + j] = s;
        }
    for (int i = 0; i < 3; ++i) {
        float s = Rtrans[(size_t)node * 3 + i];
        for (int k = 0; k < 3; ++k) s += R[i * 3 + k] * uv[node * 6 + 3 + k];
        out_trans[(size_t)node * 3 + i] = s;
    }
}

// ---------------------------------------------------------------------------
// Host side
// ---------------------------------------------------------------------------
static inline void launch_gemm(const float* A, const float* W, const float* bias,
                               float* C, int M, int K, int N, int act,
                               hipStream_t s)
{
    if (N % 128 == 0) {
        dim3 grid(M / 128, N / 128);
        gemm_wmma_tile_kernel<8><<<grid, 256, 0, s>>>(A, W, bias, C, M, K, N, act);
    } else if (N % 64 == 0) {
        dim3 grid(M / 128, N / 64);
        gemm_wmma_tile_kernel<4><<<grid, 256, 0, s>>>(A, W, bias, C, M, K, N, act);
    } else {                       // N % 48 == 0  (N = 48, 144)
        dim3 grid(M / 128, N / 48);
        gemm_wmma_tile_kernel<3><<<grid, 256, 0, s>>>(A, W, bias, C, M, K, N, act);
    }
}

extern "C" void kernel_launch(void* const* d_in, const int* in_sizes, int n_in,
                              void* d_out, int out_size, void* d_ws, size_t ws_size,
                              hipStream_t stream)
{
    // ---- inputs (setup_inputs dict order; params flattened in dict order) ----
    const float* hV_in   = (const float*)d_in[0];
    const float* hE_in   = (const float*)d_in[1];
    const int*   E_idx   = (const int*)  d_in[2];
    const float* mask_V  = (const float*)d_in[3];
    const float* mask_at = (const float*)d_in[4];
    const float* R_rot   = (const float*)d_in[5];
    const float* R_trans = (const float*)d_in[6];
    /* d_in[7] = pos (unused) */
    const float* fixedm  = (const float*)d_in[8];
    int p = 9;
    const float* W1_w  = (const float*)d_in[p++]; const float* W1_b  = (const float*)d_in[p++];
    const float* W2_w  = (const float*)d_in[p++]; const float* W2_b  = (const float*)d_in[p++];
    const float* W3_w  = (const float*)d_in[p++]; const float* W3_b  = (const float*)d_in[p++];
    const float* W11_w = (const float*)d_in[p++]; const float* W11_b = (const float*)d_in[p++];
    const float* W12_w = (const float*)d_in[p++]; const float* W12_b = (const float*)d_in[p++];
    const float* W13_w = (const float*)d_in[p++]; const float* W13_b = (const float*)d_in[p++];
    const float* ln1_g = (const float*)d_in[p++]; const float* ln1_b = (const float*)d_in[p++];
    const float* ln2_g = (const float*)d_in[p++]; const float* ln2_b = (const float*)d_in[p++];
    const float* ln3_g = (const float*)d_in[p++]; const float* ln3_b = (const float*)d_in[p++];
    const float* fin_w = (const float*)d_in[p++]; const float* fin_b = (const float*)d_in[p++];
    const float* fout_w= (const float*)d_in[p++]; const float* fout_b= (const float*)d_in[p++];
    const float* q_w   = (const float*)d_in[p++]; const float* q_b   = (const float*)d_in[p++];
    const float* kv_w  = (const float*)d_in[p++]; const float* kv_b  = (const float*)d_in[p++];
    const float* qp_w  = (const float*)d_in[p++]; const float* qp_b  = (const float*)d_in[p++];
    const float* kvp_w = (const float*)d_in[p++]; const float* kvp_b = (const float*)d_in[p++];
    const float* head_w= (const float*)d_in[p++];
    const float* out_w = (const float*)d_in[p++]; const float* out_b = (const float*)d_in[p++];
    const float* nln_g = (const float*)d_in[p++]; const float* nln_b = (const float*)d_in[p++];
    const float* t1_w  = (const float*)d_in[p++]; const float* t1_b  = (const float*)d_in[p++];
    const float* t2_w  = (const float*)d_in[p++]; const float* t2_b  = (const float*)d_in[p++];
    const float* t3_w  = (const float*)d_in[p++]; const float* t3_b  = (const float*)d_in[p++];
    const float* tln_g = (const float*)d_in[p++]; const float* tln_b = (const float*)d_in[p++];
    const float* bb_w  = (const float*)d_in[p++]; const float* bb_b  = (const float*)d_in[p++];

    // ---- output layout: h_V | h_E | new_rot | new_trans ----
    float* out      = (float*)d_out;
    float* out_hV   = out;
    float* out_hE   = out + (size_t)BN * HD;
    float* out_rot  = out_hE + (size_t)BNK * HD;
    float* out_tr   = out_rot + (size_t)BN * 9;

    // ---- workspace carve-out ----
    float* ws = (float*)d_ws;
    size_t off = 0;
    auto alloc = [&](size_t n) { float* r = ws + off; off += n; return r; };
    float* hEV   = alloc((size_t)BNK * DIN);    // also reused for q/kv/qp/kvp
    float* bufE1 = alloc((size_t)BNK * HD);
    float* bufE2 = alloc((size_t)BNK * HD);
    float* feats = alloc((size_t)BNK * 192);
    float* Rjb   = alloc((size_t)BNK * 9);
    float* tjb   = alloc((size_t)BNK * 3);
    float* hV1   = alloc((size_t)BN * HD);
    float* hV2   = alloc((size_t)BN * HD);
    float* dhbig = alloc((size_t)BN * 512);
    float* dh    = alloc((size_t)BN * HD);
    float* sums  = alloc((size_t)BN * HD);
    float* cnts  = alloc((size_t)BN);
    float* ne    = alloc((size_t)BN * HD);
    float* tb1   = alloc((size_t)BN * HD);
    float* tb2   = alloc((size_t)BN * HD);
    float* ne2   = alloc((size_t)BN * HD);
    float* uvb   = alloc((size_t)BN * 6);
    (void)ws_size; (void)in_sizes; (void)n_in; (void)out_size;

    // IPA projection outputs packed into the hEV region (64+128+48+144 = 384)
    float* qbuf   = hEV;
    float* kvbuf  = hEV + (size_t)BNK * 64;
    float* qpbuf  = hEV + (size_t)BNK * 192;
    float* kvpbuf = hEV + (size_t)BNK * 240;

    // ======================= Phase 1: first message block ====================
    build_hEV_kernel<<<BNK, DIN, 0, stream>>>(hV_in, hE_in, E_idx, hEV);
    launch_gemm(hEV,   W1_w, W1_b, bufE1, BNK, DIN, HD, 1, stream);
    launch_gemm(bufE1, W2_w, W2_b, bufE2, BNK, HD,  HD, 1, stream);
    launch_gemm(bufE2, W3_w, W3_b, bufE1, BNK, HD,  HD, 0, stream);
    node_update_kernel<<<BN, HD, 0, stream>>>(hV_in, bufE1, mask_at, ln1_g, ln1_b, hV1);

    // ======================= Phase 2: node FFN ==============================
    launch_gemm(hV1,   fin_w,  fin_b,  dhbig, BN, HD,  512, 1, stream);
    launch_gemm(dhbig, fout_w, fout_b, dh,    BN, 512, HD,  0, stream);
    add_ln_kernel<<<BN, HD, 0, stream>>>(hV1, dh, ln2_g, ln2_b, mask_V, hV2);

    // ======================= Phase 3: edge update ===========================
    build_hEV_kernel<<<BNK, DIN, 0, stream>>>(hV2, hE_in, E_idx, hEV);
    launch_gemm(hEV,   W11_w, W11_b, bufE1, BNK, DIN, HD, 1, stream);
    launch_gemm(bufE1, W12_w, W12_b, bufE2, BNK, HD,  HD, 1, stream);
    launch_gemm(bufE2, W13_w, W13_b, bufE1, BNK, HD,  HD, 0, stream);
    add_ln_kernel<<<BNK, HD, 0, stream>>>(hE_in, bufE1, ln3_g, ln3_b, nullptr, out_hE);

    // ======================= Phase 4: IPA ===================================
    gather_frames_kernel<<<(BNK + 255) / 256, 256, 0, stream>>>(E_idx, R_rot, R_trans, Rjb, tjb);
    launch_gemm(out_hE, q_w,   q_b,   qbuf,   BNK, HD, 64,  0, stream);
    launch_gemm(out_hE, kv_w,  kv_b,  kvbuf,  BNK, HD, 128, 0, stream);
    launch_gemm(out_hE, qp_w,  qp_b,  qpbuf,  BNK, HD, 48,  0, stream);
    launch_gemm(out_hE, kvp_w, kvp_b, kvpbuf, BNK, HD, 144, 0, stream);
    {
        long tq = (long)BNK * 16;
        apply_rot_kernel<<<(int)((tq + 255) / 256), 256, 0, stream>>>(qpbuf, Rjb, tjb, 16);
        long tk = (long)BNK * 48;
        apply_rot_kernel<<<(int)((tk + 255) / 256), 256, 0, stream>>>(kvpbuf, Rjb, tjb, 48);
    }
    {
        long tot = (long)BN * NH4 * NK;
        ipa_attn_kernel<<<(int)((tot + 255) / 256), 256, 0, stream>>>(
            qbuf, kvbuf, qpbuf, kvpbuf, Rjb, tjb, head_w, mask_at, feats);
    }
    launch_gemm(feats, out_w, out_b, bufE2, BNK, 192, HD, 0, stream);   // ipa_out

    // ======================= Phase 5: segment mean + transition =============
    zero_kernel<<<(int)(((long)BN * HD + 255) / 256), 256, 0, stream>>>(sums, (long)BN * HD);
    zero_kernel<<<(BN + 255) / 256, 256, 0, stream>>>(cnts, BN);
    {
        long tot = (long)BNK * HD;
        seg_accum_kernel<<<(int)((tot + 255) / 256), 256, 0, stream>>>(
            E_idx, mask_at, bufE2, sums, cnts);
    }
    seg_ln_kernel<<<BN, HD, 0, stream>>>(sums, cnts, nln_g, nln_b, ne);
    launch_gemm(ne,  t1_w, t1_b, tb1, BN, HD, HD, 2, stream);
    launch_gemm(tb1, t2_w, t2_b, tb2, BN, HD, HD, 2, stream);
    launch_gemm(tb2, t3_w, t3_b, tb1, BN, HD, HD, 0, stream);
    add_ln_kernel<<<BN, HD, 0, stream>>>(tb1, ne, tln_g, tln_b, nullptr, ne2);

    // ======================= Phase 6: backbone update + outputs =============
    bb_kernel<<<(BN * 6 + 255) / 256, 256, 0, stream>>>(ne2, mask_V, fixedm, bb_w, bb_b, uvb);
    add_elem_kernel<<<(int)(((long)BN * HD + 255) / 256), 256, 0, stream>>>(
        hV2, ne2, out_hV, (long)BN * HD);
    finalize_kernel<<<(BN + 255) / 256, 256, 0, stream>>>(uvb, R_rot, R_trans, out_rot, out_tr);
}
